// Attention_16484084483742
// MI455X (gfx1250) — compile-verified, matrix operand order checked by reading, not run on hardware
//
#include <hip/hip_runtime.h>
#include <hip/hip_bf16.h>

typedef __attribute__((ext_vector_type(16))) __bf16 v16bf;
typedef __attribute__((ext_vector_type(4)))  __bf16 v4bf;
typedef __attribute__((ext_vector_type(8)))  float  v8f;
typedef __attribute__((ext_vector_type(4)))  unsigned v4u;
typedef __attribute__((ext_vector_type(8)))  int    v8i;
typedef __attribute__((ext_vector_type(4)))  int    v4i;

#define ATT_B 64
#define ATT_N 196
#define ATT_H 12
#define ATT_D 64
#define ATT_C 768
#define QKV_C 2304
#define NPAD  224   // 196 padded to 7*32 for the P@V contraction
#define ROWS  12544 // B*N

#if __has_builtin(__builtin_amdgcn_tensor_load_to_lds)
#define HAVE_TDM 1
#pragma message("CDNA5 probe: HAVE_TDM=1 (tensor_load_to_lds builtin present)")
#if __clang_major__ >= 23
#pragma message("CDNA5 probe: using 6-arg tensor_load_to_lds")
#else
#pragma message("CDNA5 probe: using 5-arg tensor_load_to_lds")
#endif
#else
#define HAVE_TDM 0
#pragma message("CDNA5 probe: HAVE_TDM=0 (fallback cooperative staging)")
#endif

__device__ __forceinline__ void store_out(float* p, float v)  { *p = v; }
__device__ __forceinline__ void store_out(__bf16* p, float v) { *p = (__bf16)v; }

// Low 32 bits of a generic pointer to __shared__ = byte offset in LDS space.
__device__ __forceinline__ unsigned lds_off(const void* p) {
  return (unsigned)(size_t)p;
}

#if HAVE_TDM
// Issue one TDM 2D tile load: rows x cols bf16 elements from a row-major
// tensor with row stride `stride` (elements) into LDS at byte offset lds_addr,
// packed contiguously as [rows][cols].  D# layout per CDNA5 ISA 08 §8.3/8.4.
__device__ __forceinline__ void tdm_load_2d_bf16(unsigned lds_addr,
                                                 const void* gptr,
                                                 unsigned rows, unsigned cols,
                                                 unsigned stride) {
  const unsigned long long ga = (unsigned long long)(size_t)gptr;
  v4u g0;
  g0[0] = 1u;                                          // count=1, user desc
  g0[1] = lds_addr;                                    // lds_addr (bytes)
  g0[2] = (unsigned)(ga & 0xffffffffu);                // global_addr[31:0]
  g0[3] = (unsigned)((ga >> 32) & 0x1ffffffu) | (2u << 30);  // addr[56:32]|type=2
  v8i g1;
  g1[0] = (int)(1u << 16);                 // workgroup_mask=0, data_size=1 (2B)
  g1[1] = (int)(cols << 16);               // tensor_dim0[15:0]
  g1[2] = (int)(rows << 16);               // tensor_dim0[31:16]=0, tensor_dim1[15:0]
  g1[3] = (int)(cols << 16);               // tensor_dim1[31:16]=0, tile_dim0
  g1[4] = (int)rows;                       // tile_dim1 (tile_dim2=0)
  g1[5] = (int)stride;                     // tensor_dim0_stride[31:0]
  g1[6] = 0;                               // stride[47:32], dim1_stride[15:0]=0
  g1[7] = 0;
  const v4i z4 = {0, 0, 0, 0};
#if __clang_major__ >= 23
  const v8i z8 = {0, 0, 0, 0, 0, 0, 0, 0};
  __builtin_amdgcn_tensor_load_to_lds(g0, g1, z4, z4, z8, 0);
#else
  __builtin_amdgcn_tensor_load_to_lds(g0, g1, z4, z4, 0);
#endif
}
#endif

// Load a 16x32 bf16 WMMA operand fragment (A orientation) from a row-major
// [16][ld] buffer. Per ISA 7.12.2: row = lane%16; lanes 0-15 hold K {0..7,16..23},
// lanes 16-31 hold K {8..15,24..31}, two K per VGPR.
__device__ __forceinline__ v16bf frag_rowmajor_lds(const __bf16* buf, int ld) {
  const int lane = threadIdx.x & 31;
  const int r    = lane & 15;
  const int half = (lane >> 4) << 3;
  v16bf f;
#pragma unroll
  for (int i = 0; i < 8; ++i) {
    const int kb = (i < 4) ? (2 * i) : (16 + 2 * (i - 4));
    const int k  = kb + half;
    f[2 * i]     = buf[r * ld + k];
    f[2 * i + 1] = buf[r * ld + k + 1];
  }
  return f;
}

// ---------------------------------------------------------------------------
// f32 -> bf16 conversion (vectorized x4)
// ---------------------------------------------------------------------------
__global__ __launch_bounds__(256) void cvt_f32_to_bf16(
    const float* __restrict__ in, __bf16* __restrict__ out, int n4) {
  const int idx = blockIdx.x * 256 + threadIdx.x;
  if (idx < n4) {
    const float4 v = ((const float4*)in)[idx];
    v4bf o;
    o[0] = (__bf16)v.x; o[1] = (__bf16)v.y;
    o[2] = (__bf16)v.z; o[3] = (__bf16)v.w;
    ((v4bf*)out)[idx] = o;
  }
}

// ---------------------------------------------------------------------------
// C[M,N] = A[M,K] @ B[N,K]^T + bias[N], A/B bf16, fp32 WMMA accumulation.
// M%128==0, N%128==0, K%32==0.  256 threads (8 waves) own a 128x128 tile;
// wave (wr = w&1, wc = w>>1) computes a 64x32 sub-tile: 8 WMMAs per k-step
// against 6 fragment loads (10.7 MAC/LDS-byte).  K-panels (128x32 bf16) are
// staged into double-buffered LDS by the Tensor Data Mover: wave 0 issues the
// DMA for panel i+1 which overlaps the WMMA burst on panel i; in-order TDM
// completion makes s_wait_tensorcnt(0) release exactly the panel consumed.
// ---------------------------------------------------------------------------
template <typename TO>
__global__ __launch_bounds__(256) void gemm_wmma(const __bf16* __restrict__ A,
                                                 const __bf16* __restrict__ Bm,
                                                 const float* __restrict__ bias,
                                                 TO* __restrict__ Cm,
                                                 int M, int N, int K) {
  __shared__ __attribute__((aligned(16))) __bf16 As[2][128 * 32];  // 2 x 8 KB
  __shared__ __attribute__((aligned(16))) __bf16 Bs[2][128 * 32];  // 2 x 8 KB
  const int tn   = blockIdx.x * 128;
  const int tm   = blockIdx.y * 128;
  const int tid  = threadIdx.x;
  const int wave = tid >> 5;
  const int lane = tid & 31;
  const int wr   = wave & 1;   // 64-row group within the 128-row tile
  const int wc   = wave >> 1;  // 32-col group within the 128-col tile
  const int nk   = K >> 5;

  v8f acc[4][2] = {};

#if HAVE_TDM
  const bool tdm_wave = (wave == 0);
  if (tdm_wave) {
    tdm_load_2d_bf16(lds_off(As[0]), A + (size_t)tm * K, 128, 32, K);
    tdm_load_2d_bf16(lds_off(Bs[0]), Bm + (size_t)tn * K, 128, 32, K);
  }
#endif

  for (int i = 0; i < nk; ++i) {
    const int cur = i & 1;
#if HAVE_TDM
    if (tdm_wave) __builtin_amdgcn_s_wait_tensorcnt(0);
    __syncthreads();  // panel `cur` visible to all waves; prev reads retired
    if (tdm_wave && (i + 1) < nk) {  // prefetch next panel, overlaps compute
      tdm_load_2d_bf16(lds_off(As[1 - cur]),
                       A + (size_t)tm * K + (i + 1) * 32, 128, 32, K);
      tdm_load_2d_bf16(lds_off(Bs[1 - cur]),
                       Bm + (size_t)tn * K + (i + 1) * 32, 128, 32, K);
    }
#else
    for (int t = tid; t < 128 * 32; t += 256) {
      As[cur][t] = A[(size_t)(tm + (t >> 5)) * K + i * 32 + (t & 31)];
      Bs[cur][t] = Bm[(size_t)(tn + (t >> 5)) * K + i * 32 + (t & 31)];
    }
    __syncthreads();
#endif

    v16bf bf[2];
#pragma unroll
    for (int nt = 0; nt < 2; ++nt)
      bf[nt] = frag_rowmajor_lds(Bs[cur] + (wc * 32 + nt * 16) * 32, 32);
#pragma unroll
    for (int mt = 0; mt < 4; ++mt) {
      const v16bf af = frag_rowmajor_lds(As[cur] + (wr * 64 + mt * 16) * 32, 32);
#pragma unroll
      for (int nt = 0; nt < 2; ++nt)
        acc[mt][nt] = __builtin_amdgcn_wmma_f32_16x16x32_bf16(
            false, af, false, bf[nt], (short)0, acc[mt][nt], false, false);
    }
    __syncthreads();
  }

  // C fragment layout: VGPR i, lanes 0-15 -> row i, lanes 16-31 -> row i+8.
  const int halfr = (lane >> 4) << 3;
#pragma unroll
  for (int nt = 0; nt < 2; ++nt) {
    const int col  = tn + wc * 32 + nt * 16 + (lane & 15);
    const float bv = bias[col];
#pragma unroll
    for (int mt = 0; mt < 4; ++mt)
#pragma unroll
      for (int e = 0; e < 8; ++e) {
        const int row = tm + wr * 64 + mt * 16 + e + halfr;
        store_out(&Cm[(size_t)row * N + col], acc[mt][nt][e] + bv);
      }
  }
}

// ---------------------------------------------------------------------------
// Attention core: one wave per (query 16-block, head, batch).
//   V[b,h] fetched by one TDM descriptor (overlaps S/softmax compute)
//   S = Q K^T * scale  (13 col tiles x 2 k-steps of WMMA)
//   softmax rows (fp32, shfl_xor reduction inside each 16-lane half)
//   P = softmax + static_a, staged to LDS as bf16 (196 -> 224 zero pad)
//   O = P V            (7 k-steps x 4 d tiles of WMMA)
// ---------------------------------------------------------------------------
__global__ __launch_bounds__(32) void attn_kernel(const __bf16* __restrict__ qkv,
                                                  const float* __restrict__ sa,
                                                  __bf16* __restrict__ attn_out) {
  __shared__ __attribute__((aligned(16))) __bf16 Pl[16 * NPAD];     // 7 KB
  __shared__ __attribute__((aligned(16))) __bf16 Vl[NPAD * ATT_D];  // 28 KB

  const int qb   = blockIdx.x;  // 0..12 (16-row query block)
  const int h    = blockIdx.y;  // head
  const int b    = blockIdx.z;  // batch
  const int lane = threadIdx.x & 31;
  const int r    = lane & 15;
  const int half = (lane >> 4) << 3;
  const __bf16 z = (__bf16)0.0f;

  // ---- V[b,h] (196x64 bf16 tile, row stride 2304) -> LDS via TDM ----
  const __bf16* vbase = qkv + (size_t)(b * ATT_N) * QKV_C + 2 * ATT_C + h * ATT_D;
#if HAVE_TDM
  tdm_load_2d_bf16(lds_off(Vl), vbase, ATT_N, ATT_D, QKV_C);
  // zero the pad rows 196..223 with plain LDS stores (disjoint from TDM dest)
  for (int t = lane; t < (NPAD - ATT_N) * (ATT_D / 4); t += 32)
    ((unsigned long long*)(Vl + ATT_N * ATT_D))[t] = 0ull;
#else
  for (int t = lane; t < NPAD * (ATT_D / 4); t += 32) {
    const int m = t >> 4;
    const int c = (t & 15) << 2;
    unsigned long long val = 0ull;
    if (m < ATT_N)
      val = *(const unsigned long long*)(vbase + (size_t)m * QKV_C + c);
    ((unsigned long long*)Vl)[t] = val;
  }
#endif
  // ---- zero P (pad columns 208..223 must be zero) ----
  for (int t = lane; t < 16 * NPAD / 4; t += 32)
    ((unsigned long long*)Pl)[t] = 0ull;

  // ---- Q fragments (2 k-steps over D=64), invalid query rows -> 0 ----
  v16bf qf[2];
  {
    const int  qrow = qb * 16 + r;
    const bool qv   = qrow < ATT_N;
    const __bf16* qp =
        qkv + (size_t)(b * ATT_N + (qv ? qrow : 0)) * QKV_C + h * ATT_D;
#pragma unroll
    for (int kk = 0; kk < 2; ++kk)
#pragma unroll
      for (int i = 0; i < 8; ++i) {
        const int kb = (i < 4) ? (2 * i) : (16 + 2 * (i - 4));
        const int c  = kk * 32 + kb + half;
        qf[kk][2 * i]     = qv ? qp[c]     : z;
        qf[kk][2 * i + 1] = qv ? qp[c + 1] : z;
      }
  }

  // ---- S = Q K^T over 13 key tiles ----
  v8f s[13];
#pragma unroll
  for (int mt = 0; mt < 13; ++mt) {
    const int  m  = mt * 16 + r;
    const bool kv = m < ATT_N;
    const __bf16* kp =
        qkv + (size_t)(b * ATT_N + (kv ? m : 0)) * QKV_C + ATT_C + h * ATT_D;
    v16bf kf0, kf1;
#pragma unroll
    for (int i = 0; i < 8; ++i) {
      const int kb = (i < 4) ? (2 * i) : (16 + 2 * (i - 4));
      const int c  = kb + half;
      kf0[2 * i]     = kv ? kp[c]          : z;
      kf0[2 * i + 1] = kv ? kp[c + 1]      : z;
      kf1[2 * i]     = kv ? kp[32 + c]     : z;
      kf1[2 * i + 1] = kv ? kp[32 + c + 1] : z;
    }
    v8f acc = {};
    acc = __builtin_amdgcn_wmma_f32_16x16x32_bf16(false, qf[0], false, kf0,
                                                  (short)0, acc, false, false);
    acc = __builtin_amdgcn_wmma_f32_16x16x32_bf16(false, qf[1], false, kf1,
                                                  (short)0, acc, false, false);
    s[mt] = acc;
  }

  // ---- scale + mask invalid key columns (col = lane&15 per tile) ----
  const float scale = 0.125f;  // D^-0.5
#pragma unroll
  for (int mt = 0; mt < 13; ++mt) {
    const bool valid = (mt * 16 + r) < ATT_N;
#pragma unroll
    for (int i = 0; i < 8; ++i)
      s[mt][i] = valid ? s[mt][i] * scale : -1e30f;
  }

  // ---- fp32 softmax per row (row lives across 16 lanes of one half) ----
#pragma unroll
  for (int i = 0; i < 8; ++i) {
    float mx = -3.0e38f;
#pragma unroll
    for (int mt = 0; mt < 13; ++mt) mx = fmaxf(mx, s[mt][i]);
#pragma unroll
    for (int d = 1; d < 16; d <<= 1) mx = fmaxf(mx, __shfl_xor(mx, d, 32));
    float sm = 0.0f;
#pragma unroll
    for (int mt = 0; mt < 13; ++mt) {
      const float e = __expf(s[mt][i] - mx);
      s[mt][i] = e;
      sm += e;
    }
#pragma unroll
    for (int d = 1; d < 16; d <<= 1) sm += __shfl_xor(sm, d, 32);
    const float inv = 1.0f / sm;
#pragma unroll
    for (int mt = 0; mt < 13; ++mt) s[mt][i] *= inv;
  }

  __syncthreads();  // LDS zero-staging ordered before P writes

  // ---- P = softmax + static_a, write to LDS as bf16 ----
#pragma unroll
  for (int mt = 0; mt < 13; ++mt) {
    const int col = mt * 16 + r;
#pragma unroll
    for (int i = 0; i < 8; ++i) {
      const int Mloc = i + half;
      const int qg   = qb * 16 + Mloc;
      float p = s[mt][i];
      if (col < ATT_N && qg < ATT_N)
        p += sa[((size_t)h * ATT_N + qg) * ATT_N + col];
      else if (col >= ATT_N)
        p = 0.0f;
      Pl[Mloc * NPAD + col] = (__bf16)p;
    }
  }
#if HAVE_TDM
  __builtin_amdgcn_s_wait_tensorcnt(0);  // V tile landed in LDS
#endif
  __syncthreads();

  // ---- O = P V : 7 k-steps x 4 d-tiles ----
  v8f o[4] = {};
#pragma unroll
  for (int kk = 0; kk < 7; ++kk) {
    v16bf pf;
#pragma unroll
    for (int i = 0; i < 8; ++i) {
      const int kb = (i < 4) ? (2 * i) : (16 + 2 * (i - 4));
      const int k  = kk * 32 + kb + half;
      pf[2 * i]     = Pl[r * NPAD + k];
      pf[2 * i + 1] = Pl[r * NPAD + k + 1];
    }
#pragma unroll
    for (int dt = 0; dt < 4; ++dt) {
      v16bf vf;  // B operand from K-major [NPAD][64] LDS buffer
#pragma unroll
      for (int i = 0; i < 8; ++i) {
        const int kb = (i < 4) ? (2 * i) : (16 + 2 * (i - 4));
        const int K0 = kk * 32 + kb + half;
        vf[2 * i]     = Vl[K0 * ATT_D + dt * 16 + r];
        vf[2 * i + 1] = Vl[(K0 + 1) * ATT_D + dt * 16 + r];
      }
      o[dt] = __builtin_amdgcn_wmma_f32_16x16x32_bf16(
          false, pf, false, vf, (short)0, o[dt], false, false);
    }
  }

  // ---- store attention output as bf16 [B,N,H,D] == [ROWS, 768] ----
#pragma unroll
  for (int dt = 0; dt < 4; ++dt) {
    const int d = dt * 16 + r;
#pragma unroll
    for (int i = 0; i < 8; ++i) {
      const int qg = qb * 16 + i + half;
      if (qg < ATT_N)
        attn_out[(size_t)(b * ATT_N + qg) * ATT_C + h * ATT_D + d] =
            (__bf16)o[dt][i];
    }
  }
}

// ---------------------------------------------------------------------------
extern "C" void kernel_launch(void* const* d_in, const int* in_sizes, int n_in,
                              void* d_out, int out_size, void* d_ws,
                              size_t ws_size, hipStream_t stream) {
  const float* x        = (const float*)d_in[0];
  const float* Wqkv     = (const float*)d_in[1];
  const float* bqkv     = (const float*)d_in[2];
  const float* Wproj    = (const float*)d_in[3];
  const float* bproj    = (const float*)d_in[4];
  const float* static_a = (const float*)d_in[5];
  float* out = (float*)d_out;

  __bf16* qkvbuf  = (__bf16*)d_ws;                        // [12544][2304]
  __bf16* attnbuf = qkvbuf + (size_t)ROWS * QKV_C;        // [12544][768]
  __bf16* xb      = attnbuf + (size_t)ROWS * ATT_C;       // [12544][768]
  __bf16* wqkvb   = xb + (size_t)ROWS * ATT_C;            // [2304][768]
  __bf16* wprojb  = wqkvb + (size_t)QKV_C * ATT_C;        // [768][768]

  // 0) one-time bf16 downconverts (also halves GEMM HBM traffic)
  {
    const int n4x = ROWS * ATT_C / 4;
    cvt_f32_to_bf16<<<(n4x + 255) / 256, 256, 0, stream>>>(x, xb, n4x);
    const int n4q = QKV_C * ATT_C / 4;
    cvt_f32_to_bf16<<<(n4q + 255) / 256, 256, 0, stream>>>(Wqkv, wqkvb, n4q);
    const int n4p = ATT_C * ATT_C / 4;
    cvt_f32_to_bf16<<<(n4p + 255) / 256, 256, 0, stream>>>(Wproj, wprojb, n4p);
  }

  // 1) qkv = x @ Wqkv^T + bqkv   (M=12544, N=2304, K=768)
  dim3 g1(QKV_C / 128, ROWS / 128);
  gemm_wmma<__bf16><<<g1, 256, 0, stream>>>(xb, wqkvb, bqkv, qkvbuf,
                                            ROWS, QKV_C, ATT_C);

  // 2) attention core (softmax + static bias + P@V)
  dim3 g2((ATT_N + 15) / 16, ATT_H, ATT_B);
  attn_kernel<<<g2, 32, 0, stream>>>(qkvbuf, static_a, attnbuf);

  // 3) out = attn @ Wproj^T + bproj  (M=12544, N=768, K=768)
  dim3 g3(ATT_C / 128, ROWS / 128);
  gemm_wmma<float><<<g3, 256, 0, stream>>>(attnbuf, wprojb, bproj, out,
                                           ROWS, ATT_C, ATT_C);
}
